// EigenActivation_74586402062339
// MI455X (gfx1250) — matrix operand order repaired and checked by reading, not compiled, not run on hardware
//
#include <hip/hip_runtime.h>
#include <hip/hip_bf16.h>

// ---------------------------------------------------------------------------
// EigenActivation (rectify, eps) without an eigensolver:
//   f(X) = 0.5 * ( X + eps*I + |X - eps*I| ),  |M| = M * sign(M)
// sign(M) via Newton-Schulz:  S <- 0.5 * S * (3I - S*S),  S0 = M / ||M||_F
// All GEMMs: 64x64 via v_wmma_f32_16x16x32_bf16 (bf16 operands, f32 accum).
// One workgroup (8 wave32) per matrix; everything staged in LDS.
//
// Structure exploited (all iterates are symmetric polynomials of M):
//  * B fragments read row-major via B[k][n] == B[n][k] -> ds_load_b128 pairs.
//  * Epilogues store the (symmetric) result TRANSPOSED: a lane's 8 C-values
//    (rows crow..crow+7, fixed col) become 8 contiguous elements at
//    D[col*64 + crow] -> one 16B packed store instead of 8 scalar b16 stores.
//  * NS epilogues fused into the accumulator store; S ping/pong buffer keeps
//    2 barriers/iteration.
// ---------------------------------------------------------------------------

typedef __attribute__((ext_vector_type(16))) __bf16 v16bf;
typedef __attribute__((ext_vector_type(8)))  __bf16 v8bf;
typedef __attribute__((ext_vector_type(8)))  float  v8f;

#define MAT_N   64
#define MAT_NN  4096          // 64*64
#define NS_ITER 24
#define EPS_F   1e-5f

// Epilogue modes
#define EPI_W     1   // Dbf^T = bf16(3I - C)          (W = 3I - S*S)
#define EPI_HALF  2   // Dbf^T = bf16(0.5 * C)         (S' = 0.5 * S*W)
#define EPI_F32   0   // Df^T  = C                     (final Y*sign, f32)

// C = A(64x64 bf16) * B(64x64 bf16); A,B symmetric, result stored transposed.
// 8 waves -> 2 of 16 output 16x16 tiles each; K=64 as two K=32 WMMA steps.
template <int MODE>
__device__ __forceinline__ void gemm64_sym(const __bf16* __restrict__ Abuf,
                                           const __bf16* __restrict__ Bbuf,
                                           __bf16* __restrict__ Dbf,
                                           float*  __restrict__ Df) {
  const int lane = threadIdx.x & 31;
  const int wave = threadIdx.x >> 5;
  const int mlo  = lane & 15;    // M (A rows) / N (B cols, C cols) within tile
  const int hi   = lane >> 4;    // lane-half selector

#pragma unroll
  for (int tt = 0; tt < 2; ++tt) {            // straight-line: no EXEC masking
    const int t  = wave + 8 * tt;
    const int ti = t >> 2;                    // tile row
    const int tj = t & 3;                     // tile col (same for both tt)
    v8f c = {};
#pragma unroll
    for (int kc = 0; kc < MAT_N; kc += 32) {
      v16bf a, b;
      // --- A fragment: 16x32 bf16 (ISA 7.12.2 16-bit A layout) ---
      // lane<16: K in {0..7}u{16..23}; lane>=16: +8. Two 16B runs -> b128 x2.
      {
        const __bf16* rowp = Abuf + (ti * 16 + mlo) * MAT_N + kc + (hi ? 8 : 0);
#pragma unroll
        for (int r = 0; r < 8; ++r) {
          const int kb = ((r & 4) ? 16 : 0) + 2 * (r & 3);
          a[2 * r]     = rowp[kb];
          a[2 * r + 1] = rowp[kb + 1];
        }
      }
      // --- B fragment: 32x16, loaded via symmetry B[k][n] = B[n][k] ---
      // lane<16: K in {0..7}u{16..23}; lane>=16: +8. Two 16B runs -> b128 x2.
      {
        const __bf16* rowp = Bbuf + (tj * 16 + mlo) * MAT_N + kc + (hi ? 8 : 0);
#pragma unroll
        for (int r = 0; r < 8; ++r) {
          const int kb = ((r & 4) ? 16 : 0) + 2 * (r & 3);
          b[2 * r]     = rowp[kb];
          b[2 * r + 1] = rowp[kb + 1];
        }
      }
      c = __builtin_amdgcn_wmma_f32_16x16x32_bf16(
              /*neg_a=*/false, a, /*neg_b=*/false, b,
              /*c_mod=*/(short)0, c, /*reuse_a=*/false, /*reuse_b=*/false);
    }
    // --- fused epilogue, TRANSPOSED store (result symmetric) ---
    // C/D layout: VGPR p -> row crow+p, col ccol. Transposed: 8 contiguous
    // elements at D[ccol*64 + crow], 16B-aligned (crow multiple of 8).
    const int crow = ti * 16 + hi * 8;
    const int ccol = tj * 16 + mlo;
    if (MODE == EPI_W) {
      v8bf d;
#pragma unroll
      for (int p = 0; p < 8; ++p) {
        const float w = (((crow + p) == ccol) ? 3.0f : 0.0f) - c[p];
        d[p] = (__bf16)w;
      }
      *(v8bf*)(Dbf + ccol * MAT_N + crow) = d;
    } else if (MODE == EPI_HALF) {
      v8bf d;
#pragma unroll
      for (int p = 0; p < 8; ++p) d[p] = (__bf16)(0.5f * c[p]);
      *(v8bf*)(Dbf + ccol * MAT_N + crow) = d;
    } else {
      float4 lo = make_float4(c[0], c[1], c[2], c[3]);
      float4 hi4 = make_float4(c[4], c[5], c[6], c[7]);
      float* dp = Df + ccol * MAT_N + crow;
      *(float4*)dp       = lo;
      *(float4*)(dp + 4) = hi4;
    }
  }
}

__global__ __launch_bounds__(256)
void eigen_activation_rectify_kernel(const float* __restrict__ x,
                                     float* __restrict__ out) {
  __shared__ __align__(16) float  sX[MAT_NN];   // original X (f32)
  __shared__ __align__(16) float  sT[MAT_NN];   // final f32 GEMM result
  __shared__ __align__(16) __bf16 sS0[MAT_NN];  // sign iterate ping
  __shared__ __align__(16) __bf16 sS1[MAT_NN];  // sign iterate pong
  __shared__ __align__(16) __bf16 sW[MAT_NN];   // W = 3I - S*S
  __shared__ float snorm;

  const int tid  = threadIdx.x;
  const int lane = tid & 31;
  const size_t base = (size_t)blockIdx.x * MAT_NN;
  const float* xb = x + base;

  if (tid == 0) snorm = 0.0f;

  // ---- Load X (float4 vectorized: 16 floats/thread) ----
#pragma unroll
  for (int j = 0; j < 4; ++j) {
    const int e = (tid + 256 * j) * 4;
    *(float4*)&sX[e] = *(const float4*)(xb + e);
  }
  __syncthreads();

  // ---- ||X - eps*I||_F^2 : wave shfl-reduce + one ds_add_f32 per wave ----
  float part = 0.0f;
#pragma unroll
  for (int j = 0; j < 16; ++j) {
    const int i = tid + 256 * j;
    const int r = i >> 6, cc = i & 63;
    const float m = sX[i] - ((r == cc) ? EPS_F : 0.0f);
    part += m * m;
  }
#pragma unroll
  for (int off = 16; off > 0; off >>= 1)
    part += __shfl_down(part, off, 32);
  if (lane == 0) atomicAdd(&snorm, part);
  __syncthreads();

  const float fnorm = sqrtf(snorm + 1e-30f);
  const float rn    = 1.0f / fnorm;

  // ---- S0 = (X - eps*I) / ||M||_F  (bf16) ----
#pragma unroll
  for (int j = 0; j < 16; ++j) {
    const int i = tid + 256 * j;
    const int r = i >> 6, cc = i & 63;
    const float m = sX[i] - ((r == cc) ? EPS_F : 0.0f);
    sS0[i] = (__bf16)(m * rn);
  }
  __syncthreads();

  // ---- Newton-Schulz with fused transposed epilogues + S ping/pong ----
  __bf16* Sc = sS0;
  __bf16* Sn = sS1;
  for (int it = 0; it < NS_ITER; ++it) {
    gemm64_sym<EPI_W>(Sc, Sc, sW, nullptr);     // W = 3I - S*S (bf16)
    __syncthreads();
    gemm64_sym<EPI_HALF>(Sc, sW, Sn, nullptr);  // S' = 0.5 * S*W (bf16)
    __syncthreads();
    __bf16* tmp = Sc; Sc = Sn; Sn = tmp;
  }

  // ---- |M| = fnorm * (Y * sign),  Y = (X - eps*I)/fnorm (reuse sW) ----
#pragma unroll
  for (int j = 0; j < 16; ++j) {
    const int i = tid + 256 * j;
    const int r = i >> 6, cc = i & 63;
    const float m = sX[i] - ((r == cc) ? EPS_F : 0.0f);
    sW[i] = (__bf16)(m * rn);
  }
  __syncthreads();
  gemm64_sym<EPI_F32>(sW, Sc, nullptr, sT);     // T = Y * sign (f32)
  __syncthreads();

  // ---- out = 0.5 * (X + eps*I + fnorm*T), float4-coalesced ----
  float* ob = out + base;
#pragma unroll
  for (int j = 0; j < 4; ++j) {
    const int e = (tid + 256 * j) * 4;
    const int r = e >> 6;
    float4 xv = *(const float4*)&sX[e];
    float4 tv = *(const float4*)&sT[e];
    float4 o;
    o.x = 0.5f * (xv.x + ((r == ((e + 0) & 63)) ? EPS_F : 0.0f) + fnorm * tv.x);
    o.y = 0.5f * (xv.y + ((r == ((e + 1) & 63)) ? EPS_F : 0.0f) + fnorm * tv.y);
    o.z = 0.5f * (xv.z + ((r == ((e + 2) & 63)) ? EPS_F : 0.0f) + fnorm * tv.z);
    o.w = 0.5f * (xv.w + ((r == ((e + 3) & 63)) ? EPS_F : 0.0f) + fnorm * tv.w);
    *(float4*)(ob + e) = o;
  }
}

extern "C" void kernel_launch(void* const* d_in, const int* in_sizes, int n_in,
                              void* d_out, int out_size, void* d_ws, size_t ws_size,
                              hipStream_t stream) {
  const float* x  = (const float*)d_in[0];
  float* out      = (float*)d_out;
  const int nbat  = in_sizes[0] / MAT_NN;   // B = 8192
  eigen_activation_rectify_kernel<<<nbat, 256, 0, stream>>>(x, out);
}
